// Model_CN_81784767250670
// MI455X (gfx1250) — matrix operand order repaired and checked by reading, not compiled, not run on hardware
//
#include <hip/hip_runtime.h>
#include <cstdint>
#include <cstddef>

typedef __attribute__((ext_vector_type(16))) _Float16 v16h;
typedef __attribute__((ext_vector_type(8)))  float    v8f;

#define BN_EPS 1e-5f

// =====================================================================
// Pack GAT weights [Wl(din x 64) | Wr(din x 64) | Wres(din x 16)] into
// WMMA B-fragment layout (ISA 7.12.2, 16-bit B 32x16 per fragment):
//   fragment(kt,nt), lane l, elem e:  K = kt*32 + (l>=16 ? 8:0) + (e&7) + (e>=8 ? 16:0)
//                                     Ncol = nt*16 + (l & 15)
// Also builds the 144-wide bias vector [bl | br | g_bias].
// =====================================================================
__global__ void __launch_bounds__(256) pack_gat_weights(
    const float* __restrict__ Wl, const float* __restrict__ bl,
    const float* __restrict__ Wr, const float* __restrict__ br,
    const float* __restrict__ Wres, const float* __restrict__ gbias,
    int din, int nkt, _Float16* __restrict__ Bp, float* __restrict__ bias)
{
    int tid = blockIdx.x * blockDim.x + threadIdx.x;
    int total = nkt * 9 * 32 * 16;
    if (tid < total) {
        int e    = tid & 15;
        int lane = (tid >> 4) & 31;
        int r    = tid >> 9;          // kt*9 + nt
        int nt   = r % 9;
        int kt   = r / 9;
        int K    = kt * 32 + ((lane & 16) ? 8 : 0) + (e & 7) + ((e & 8) ? 16 : 0);
        int col  = nt * 16 + (lane & 15);
        float v = 0.0f;
        if (K < din) {
            if (col < 64)       v = Wl[K * 64 + col];
            else if (col < 128) v = Wr[K * 64 + (col - 64)];
            else                v = Wres[K * 16 + (col - 128)];
        }
        Bp[tid] = (_Float16)v;        // tid == ((kt*9+nt)*32+lane)*16+e
    }
    if (tid < 144) {
        bias[tid] = (tid < 64) ? bl[tid]
                  : (tid < 128) ? br[tid - 64]
                  : gbias[tid - 128];
    }
}

// =====================================================================
// WMMA GEMM: Y[M x 144] = A[M x KDIM] (f32 -> f16) @ Bpacked + bias.
// One 16x16 tile per wave, 4 waves/block; K tiled by 32, fully unrolled
// (NKT/KDIM compile-time so all bounds checks fold and A loads are b128).
// A-fragment (ISA 7.12.2, 16-bit A 16x32):
//   lanes 0-15 : row M=l,   elems = K{kb..kb+7} ++ K{kb+16..kb+23}, kb=kt*32
//   lanes 16-31: row M=l&15, same with kb=kt*32+8
// D layout: VGPR j -> row (j | j+8), col = lane&15.
// =====================================================================
template<int NKT, int KDIM>
__global__ void __launch_bounds__(128) gemm_wmma_f16(
    const float* __restrict__ A, int lda, int nrows,
    const _Float16* __restrict__ Bp, const float* __restrict__ bias,
    float* __restrict__ Y, int ldy, int ntilesTotal)
{
    int wave = threadIdx.x >> 5;
    int tile = blockIdx.x * 4 + wave;
    if (tile >= ntilesTotal) return;            // uniform per wave: EXEC stays all-ones
    int mt = tile / 9;
    int nt = tile % 9;
    int lane = threadIdx.x & 31;
    int l15  = lane & 15;
    int row  = mt * 16 + l15;
    int ar   = (row < nrows) ? row : (nrows - 1);
    int khalf = (lane & 16) ? 8 : 0;
    const float* __restrict__ arow = A + (size_t)ar * lda;

    v8f acc = {};
#pragma unroll
    for (int kt = 0; kt < NKT; ++kt) {
        const int kb = kt * 32;
        v16h a;
        // first K-half: K = kb+khalf .. +7  (valid whenever kb+16 <= KDIM)
        float4 f0 = *(const float4*)(arow + kb + khalf);
        float4 f1 = *(const float4*)(arow + kb + khalf + 4);
        a[0] = (_Float16)f0.x; a[1] = (_Float16)f0.y;
        a[2] = (_Float16)f0.z; a[3] = (_Float16)f0.w;
        a[4] = (_Float16)f1.x; a[5] = (_Float16)f1.y;
        a[6] = (_Float16)f1.z; a[7] = (_Float16)f1.w;
        // second K-half: K = kb+16+khalf .. +7 (valid iff kb+32 <= KDIM)
        if (kb + 32 <= KDIM) {
            float4 f2 = *(const float4*)(arow + kb + 16 + khalf);
            float4 f3 = *(const float4*)(arow + kb + 16 + khalf + 4);
            a[8]  = (_Float16)f2.x; a[9]  = (_Float16)f2.y;
            a[10] = (_Float16)f2.z; a[11] = (_Float16)f2.w;
            a[12] = (_Float16)f3.x; a[13] = (_Float16)f3.y;
            a[14] = (_Float16)f3.z; a[15] = (_Float16)f3.w;
        } else {
#pragma unroll
            for (int j = 8; j < 16; ++j) a[j] = (_Float16)0.0f;
        }
        v16h b = *(const v16h*)(Bp + ((size_t)(kt * 9 + nt) * 512) + lane * 16);
        acc = __builtin_amdgcn_wmma_f32_16x16x32_f16(
                  false, a, false, b, (short)0, acc, false, false);
    }

    int col = nt * 16 + l15;
    float bv = bias[col];
    int rbase = mt * 16 + ((lane & 16) ? 8 : 0);
#pragma unroll
    for (int j = 0; j < 8; ++j) {
        int rr = rbase + j;
        if (rr < nrows) Y[(size_t)rr * ldy + col] = acc[j] + bv;
    }
}

// =====================================================================
// Order-preserving float<->uint encoding for atomic max on f32 scores.
// =====================================================================
__device__ __forceinline__ unsigned enc_f32(float f) {
    unsigned u = __float_as_uint(f);
    return (u & 0x80000000u) ? ~u : (u | 0x80000000u);
}
__device__ __forceinline__ float dec_f32(unsigned enc) {
    unsigned u = (enc & 0x80000000u) ? (enc ^ 0x80000000u) : ~enc;
    return __uint_as_float(u);
}

// =====================================================================
// Edge pass 1: score[e][h] = sum_c att[h][c]*leaky(xl[src]+xr[dst],0.2)
// and atomic running max per (dst,h). float4-vectorized row gathers.
// =====================================================================
__global__ void __launch_bounds__(256) edge_score_max(
    const int* __restrict__ esrc, const int* __restrict__ edst, int E, int EP,
    const float* __restrict__ Y, const float* __restrict__ att,
    float* __restrict__ score, unsigned* __restrict__ maxenc)
{
    int e = blockIdx.x * blockDim.x + threadIdx.x;
    if (e >= EP) return;
    int s = (e < E) ? esrc[e] : (e - E);       // appended self-loops
    int d = (e < E) ? edst[e] : (e - E);
    const float* xl = Y + (size_t)s * 144;           // cols 0..63
    const float* xr = Y + (size_t)d * 144 + 64;      // cols 64..127
    float sc4[4];
#pragma unroll
    for (int h = 0; h < 4; ++h) {
        float sc = 0.0f;
#pragma unroll
        for (int q = 0; q < 4; ++q) {
            float4 va = *(const float4*)(xl + h * 16 + q * 4);
            float4 vb = *(const float4*)(xr + h * 16 + q * 4);
            float4 vw = *(const float4*)(att + h * 16 + q * 4);
            float t;
            t = va.x + vb.x; t = (t > 0.0f) ? t : 0.2f * t; sc += vw.x * t;
            t = va.y + vb.y; t = (t > 0.0f) ? t : 0.2f * t; sc += vw.y * t;
            t = va.z + vb.z; t = (t > 0.0f) ? t : 0.2f * t; sc += vw.z * t;
            t = va.w + vb.w; t = (t > 0.0f) ? t : 0.2f * t; sc += vw.w * t;
        }
        sc4[h] = sc;
    }
#pragma unroll
    for (int h = 0; h < 4; ++h) {
        score[(size_t)e * 4 + h] = sc4[h];
        atomicMax(&maxenc[(size_t)d * 4 + h], enc_f32(sc4[h]));
    }
}

// =====================================================================
// Edge pass 2: ex = exp(score - max[dst]); expsum[dst][h] += ex;
//              agg[dst][h][c] += ex * xl[src][h][c]
// =====================================================================
__global__ void __launch_bounds__(256) edge_exp_agg(
    const int* __restrict__ esrc, const int* __restrict__ edst, int E, int EP,
    const float* __restrict__ Y, const float* __restrict__ score,
    const unsigned* __restrict__ maxenc,
    float* __restrict__ expsum, float* __restrict__ agg)
{
    int e = blockIdx.x * blockDim.x + threadIdx.x;
    if (e >= EP) return;
    int s = (e < E) ? esrc[e] : (e - E);
    int d = (e < E) ? edst[e] : (e - E);
    const float* xl = Y + (size_t)s * 144;
    float ex[4];
#pragma unroll
    for (int h = 0; h < 4; ++h) {
        float m = dec_f32(maxenc[(size_t)d * 4 + h]);
        ex[h] = __expf(score[(size_t)e * 4 + h] - m);
        atomicAdd(&expsum[(size_t)d * 4 + h], ex[h]);
    }
#pragma unroll
    for (int h = 0; h < 4; ++h) {
#pragma unroll
        for (int q = 0; q < 4; ++q) {
            float4 v = *(const float4*)(xl + h * 16 + q * 4);
            float* ap = agg + (size_t)d * 64 + h * 16 + q * 4;
            atomicAdd(ap + 0, ex[h] * v.x);
            atomicAdd(ap + 1, ex[h] * v.y);
            atomicAdd(ap + 2, ex[h] * v.z);
            atomicAdd(ap + 3, ex[h] * v.w);
        }
    }
}

// =====================================================================
// Node combine: out = mean_h(agg/expsum) + res(+g_bias already in Y);
// then eval-mode BN and PReLU -> write into xcat[:, coloff..coloff+15]
// =====================================================================
__global__ void __launch_bounds__(256) node_combine(
    int N, const float* __restrict__ agg, const float* __restrict__ expsum,
    const float* __restrict__ Y,
    const float* __restrict__ bng, const float* __restrict__ bnb,
    const float* __restrict__ bnm, const float* __restrict__ bnv,
    const float* __restrict__ pra, float* __restrict__ xcat, int coloff)
{
    int t = blockIdx.x * blockDim.x + threadIdx.x;
    if (t >= N * 16) return;
    int n = t >> 4;
    int c = t & 15;
    float m = 0.0f;
#pragma unroll
    for (int h = 0; h < 4; ++h)
        m += agg[(size_t)n * 64 + h * 16 + c] / expsum[(size_t)n * 4 + h];
    m *= 0.25f;
    float out = m + Y[(size_t)n * 144 + 128 + c];        // residual + bias
    out = (out - bnm[c]) * rsqrtf(bnv[c] + BN_EPS) * bng[c] + bnb[c];
    out = (out > 0.0f) ? out : pra[c] * out;
    xcat[(size_t)n * 32 + coloff + c] = out;
}

// =====================================================================
// Label-edge MLP: feat(86) = [x_all[s] | x_all[d] | attr]; BN0; fc1+BN1+
// leaky(0.01); fc2+BN2+leaky; fc3 -> logits(2). One thread per edge.
// =====================================================================
__global__ void __launch_bounds__(256) mlp_edge_kernel(
    int LE, const int* __restrict__ lsrc, const int* __restrict__ ldst,
    const float* __restrict__ xcat, const float* __restrict__ eattr,
    const float* __restrict__ mn0g, const float* __restrict__ mn0b,
    const float* __restrict__ mn0m, const float* __restrict__ mn0v,
    const float* __restrict__ fc1W, const float* __restrict__ fc1b,
    const float* __restrict__ mn1g, const float* __restrict__ mn1b,
    const float* __restrict__ mn1m, const float* __restrict__ mn1v,
    const float* __restrict__ fc2W, const float* __restrict__ fc2b,
    const float* __restrict__ mn2g, const float* __restrict__ mn2b,
    const float* __restrict__ mn2m, const float* __restrict__ mn2v,
    const float* __restrict__ fc3W, const float* __restrict__ fc3b,
    float* __restrict__ featOut, float* __restrict__ logitsOut)
{
    int e = blockIdx.x * blockDim.x + threadIdx.x;
    if (e >= LE) return;
    int s = lsrc[e], d = ldst[e];
    const float* xs = xcat + (size_t)s * 32;
    const float* xd = xcat + (size_t)d * 32;
    const float* at = eattr + (size_t)e * 22;
    float* fo = featOut + (size_t)e * 86;

    float h1[32];
#pragma unroll
    for (int j = 0; j < 32; ++j) h1[j] = fc1b[j];

    for (int i = 0; i < 86; ++i) {
        float f = (i < 32) ? xs[i] : (i < 64) ? xd[i - 32] : at[i - 64];
        fo[i] = f;                                      // raw feat output
        float fn = (f - mn0m[i]) * rsqrtf(mn0v[i] + BN_EPS) * mn0g[i] + mn0b[i];
#pragma unroll
        for (int j = 0; j < 32; ++j) h1[j] += fn * fc1W[i * 32 + j];
    }

    float h2[8];
#pragma unroll
    for (int k = 0; k < 8; ++k) h2[k] = fc2b[k];
#pragma unroll
    for (int j = 0; j < 32; ++j) {
        float v = (h1[j] - mn1m[j]) * rsqrtf(mn1v[j] + BN_EPS) * mn1g[j] + mn1b[j];
        v = (v > 0.0f) ? v : 0.01f * v;
#pragma unroll
        for (int k = 0; k < 8; ++k) h2[k] += v * fc2W[j * 8 + k];
    }

    float l0 = fc3b[0], l1 = fc3b[1];
#pragma unroll
    for (int k = 0; k < 8; ++k) {
        float v = (h2[k] - mn2m[k]) * rsqrtf(mn2v[k] + BN_EPS) * mn2g[k] + mn2b[k];
        v = (v > 0.0f) ? v : 0.01f * v;
        l0 += v * fc3W[k * 2 + 0];
        l1 += v * fc3W[k * 2 + 1];
    }
    logitsOut[(size_t)e * 2 + 0] = l0;
    logitsOut[(size_t)e * 2 + 1] = l1;
}

// =====================================================================
// Host-side orchestration
// =====================================================================
extern "C" void kernel_launch(void* const* d_in, const int* in_sizes, int n_in,
                              void* d_out, int out_size, void* d_ws, size_t ws_size,
                              hipStream_t stream) {
    (void)n_in; (void)out_size; (void)ws_size;

    const int N  = in_sizes[0] / 128;   // 100000
    const int LE = in_sizes[1] / 2;     // 400000
    const int E  = in_sizes[2] / 2;     // 1600000
    const int EP = E + N;               // with self-loops

    const float* x     = (const float*)d_in[0];
    const int*   lsrc  = (const int*)d_in[1];
    const int*   ldst  = lsrc + LE;
    const int*   esrc  = (const int*)d_in[2];
    const int*   edst  = esrc + E;
    const float* eattr = (const float*)d_in[3];

    // GAT layer params (dict order)
    const float* g1_Wl  = (const float*)d_in[4];
    const float* g1_bl  = (const float*)d_in[5];
    const float* g1_Wr  = (const float*)d_in[6];
    const float* g1_br  = (const float*)d_in[7];
    const float* g1_att = (const float*)d_in[8];
    const float* g1_Wres= (const float*)d_in[9];
    const float* g1_bias= (const float*)d_in[10];
    const float* bn1g = (const float*)d_in[11]; const float* bn1b = (const float*)d_in[12];
    const float* bn1m = (const float*)d_in[13]; const float* bn1v = (const float*)d_in[14];
    const float* pr1a = (const float*)d_in[15];
    const float* g2_Wl  = (const float*)d_in[16];
    const float* g2_bl  = (const float*)d_in[17];
    const float* g2_Wr  = (const float*)d_in[18];
    const float* g2_br  = (const float*)d_in[19];
    const float* g2_att = (const float*)d_in[20];
    const float* g2_Wres= (const float*)d_in[21];
    const float* g2_bias= (const float*)d_in[22];
    const float* bn2g = (const float*)d_in[23]; const float* bn2b = (const float*)d_in[24];
    const float* bn2m = (const float*)d_in[25]; const float* bn2v = (const float*)d_in[26];
    const float* pr2a = (const float*)d_in[27];
    const float* mn0g = (const float*)d_in[28]; const float* mn0b = (const float*)d_in[29];
    const float* mn0m = (const float*)d_in[30]; const float* mn0v = (const float*)d_in[31];
    const float* fc1W = (const float*)d_in[32]; const float* fc1b = (const float*)d_in[33];
    const float* mn1g = (const float*)d_in[34]; const float* mn1b = (const float*)d_in[35];
    const float* mn1m = (const float*)d_in[36]; const float* mn1v = (const float*)d_in[37];
    const float* fc2W = (const float*)d_in[38]; const float* fc2b = (const float*)d_in[39];
    const float* mn2g = (const float*)d_in[40]; const float* mn2b = (const float*)d_in[41];
    const float* mn2m = (const float*)d_in[42]; const float* mn2v = (const float*)d_in[43];
    const float* fc3W = (const float*)d_in[44]; const float* fc3b = (const float*)d_in[45];

    // ---- workspace carving (256B aligned) ----
    char* p = (char*)d_ws;
    auto carve = [&p](size_t bytes) -> void* {
        void* r = (void*)p;
        p += (bytes + 255) & ~(size_t)255;
        return r;
    };
    _Float16* w1p   = (_Float16*)carve((size_t)4 * 9 * 512 * sizeof(_Float16));
    float*    bias1 = (float*)   carve(144 * sizeof(float));
    _Float16* w2p   = (_Float16*)carve((size_t)1 * 9 * 512 * sizeof(_Float16));
    float*    bias2 = (float*)   carve(144 * sizeof(float));
    float*    y1    = (float*)   carve((size_t)N * 144 * sizeof(float));
    float*    y2    = (float*)   carve((size_t)N * 144 * sizeof(float));
    float*    xcat  = (float*)   carve((size_t)N * 32 * sizeof(float));
    unsigned* mxe   = (unsigned*)carve((size_t)N * 4 * sizeof(unsigned));
    float*    esum  = (float*)   carve((size_t)N * 4 * sizeof(float));
    float*    agg   = (float*)   carve((size_t)N * 64 * sizeof(float));
    float*    score = (float*)   carve((size_t)EP * 4 * sizeof(float));

    float* featOut   = (float*)d_out;
    float* logitsOut = featOut + (size_t)LE * 86;

    const int mtiles = (N + 15) / 16;
    const int ntilesTotal = mtiles * 9;                 // 144 cols = 9 tiles
    const int gemmBlocks = (ntilesTotal + 3) / 4;       // 4 waves/block
    const int edgeBlocks = (EP + 255) / 256;
    const int nodeBlocks = (N * 16 + 255) / 256;

    // ---- pack weights ----
    pack_gat_weights<<<(4 * 9 * 512 + 255) / 256, 256, 0, stream>>>(
        g1_Wl, g1_bl, g1_Wr, g1_br, g1_Wres, g1_bias, 128, 4, w1p, bias1);
    pack_gat_weights<<<(1 * 9 * 512 + 255) / 256, 256, 0, stream>>>(
        g2_Wl, g2_bl, g2_Wr, g2_br, g2_Wres, g2_bias, 16, 1, w2p, bias2);

    // ================= Layer 1 =================
    gemm_wmma_f16<4, 128><<<gemmBlocks, 128, 0, stream>>>(
        x, 128, N, w1p, bias1, y1, 144, ntilesTotal);

    hipMemsetAsync(mxe,  0, (size_t)N * 4 * sizeof(unsigned), stream);
    hipMemsetAsync(esum, 0, (size_t)N * 4 * sizeof(float), stream);
    hipMemsetAsync(agg,  0, (size_t)N * 64 * sizeof(float), stream);

    edge_score_max<<<edgeBlocks, 256, 0, stream>>>(esrc, edst, E, EP, y1, g1_att, score, mxe);
    edge_exp_agg  <<<edgeBlocks, 256, 0, stream>>>(esrc, edst, E, EP, y1, score, mxe, esum, agg);
    node_combine  <<<nodeBlocks, 256, 0, stream>>>(N, agg, esum, y1,
                                                   bn1g, bn1b, bn1m, bn1v, pr1a, xcat, 0);

    // ================= Layer 2 =================
    gemm_wmma_f16<1, 16><<<gemmBlocks, 128, 0, stream>>>(
        xcat, 32, N, w2p, bias2, y2, 144, ntilesTotal);

    hipMemsetAsync(mxe,  0, (size_t)N * 4 * sizeof(unsigned), stream);
    hipMemsetAsync(esum, 0, (size_t)N * 4 * sizeof(float), stream);
    hipMemsetAsync(agg,  0, (size_t)N * 64 * sizeof(float), stream);

    edge_score_max<<<edgeBlocks, 256, 0, stream>>>(esrc, edst, E, EP, y2, g2_att, score, mxe);
    edge_exp_agg  <<<edgeBlocks, 256, 0, stream>>>(esrc, edst, E, EP, y2, score, mxe, esum, agg);
    node_combine  <<<nodeBlocks, 256, 0, stream>>>(N, agg, esum, y2,
                                                   bn2g, bn2b, bn2m, bn2v, pr2a, xcat, 16);

    // ================= Label-edge MLP =================
    mlp_edge_kernel<<<(LE + 255) / 256, 256, 0, stream>>>(
        LE, lsrc, ldst, xcat, eattr,
        mn0g, mn0b, mn0m, mn0v, fc1W, fc1b,
        mn1g, mn1b, mn1m, mn1v, fc2W, fc2b,
        mn2g, mn2b, mn2m, mn2v, fc3W, fc3b,
        featOut, logitsOut);
}
